// SplitMLP_79766132621944
// MI455X (gfx1250) — matrix operand order repaired and checked by reading, not compiled, not run on hardware
//
#include <hip/hip_runtime.h>

// SplitMLP: B=128, C=16, V=32, H=64, O=4, G=10000
#define Bb   128
#define Cc   16
#define Vv   32
#define Hh   64
#define Oo   4
#define Gg   10000
#define K1   48        // real fc1 K (C+V); padded to 64
#define XK   72        // LDS row stride (bf16) for X and W  -> bank-conflict free A loads
#define HS   66        // LDS row stride (f32) for hidden buffer
#define W2S  68        // LDS row stride (f32) for padded W2 (16B-aligned rows, conflict-free reads)

typedef __attribute__((ext_vector_type(16))) __bf16 v16bf;
typedef __attribute__((ext_vector_type(8)))  float  v8f;
typedef __attribute__((ext_vector_type(2)))  float  v2f;
typedef __attribute__((ext_vector_type(4)))  unsigned int u32x4;
typedef __attribute__((ext_vector_type(4)))  int i32x4;

#if defined(__gfx1250__) && __has_builtin(__builtin_amdgcn_global_load_async_to_lds_b128)
#define HAVE_ASYNC_LDS 1
#define AS1 __attribute__((address_space(1)))
#define AS3 __attribute__((address_space(3)))
#else
#define HAVE_ASYNC_LDS 0
#endif

union Frag16 { v16bf v; u32x4 u[2]; };

__device__ __forceinline__ void ld16f(float* dst, const float* __restrict__ src) {
  const float4* s = reinterpret_cast<const float4*>(src);
#pragma unroll
  for (int i = 0; i < 4; ++i) {
    float4 q = s[i];
    dst[4*i+0] = q.x; dst[4*i+1] = q.y; dst[4*i+2] = q.z; dst[4*i+3] = q.w;
  }
}

__global__ __launch_bounds__(256) void splitmlp_wmma(
    const float* __restrict__ day,    const float* __restrict__ items,
    const float* __restrict__ W1_day, const float* __restrict__ W1_var,
    const float* __restrict__ b1,     const float* __restrict__ W2,
    const float* __restrict__ b2,     float* __restrict__ out)
{
  __shared__ __align__(16) __bf16 Xh[Bb * XK];   // activation hi (bf16), K-padded
  __shared__ __align__(16) __bf16 Xl[Bb * XK];   // activation lo (bf16)
  __shared__ __align__(16) __bf16 Wh[Hh * XK];   // fc1 weight hi, N-major [n][k]
  __shared__ __align__(16) __bf16 Wl[Hh * XK];   // fc1 weight lo
  __shared__ __align__(16) float  hbuf[Bb * HS]; // ReLU(fc1) fp32
  __shared__ __align__(16) float  w2s[16 * W2S]; // fc2 weight, N padded 4->16
  __shared__ __align__(16) float  b1s[Hh];
  __shared__ __align__(16) float  b2s[Oo];

  const int g    = blockIdx.x;
  const int t    = threadIdx.x;     // 0..255 (8 wave32)
  const int lane = t & 31;
  const int wave = t >> 5;          // M-tile index 0..7
  const int hsel = lane >> 4;       // half-wave select
  const int l15  = lane & 15;

  // ---- Async DMA the pure-fp32 tiles (W2 rows, b1, b2) global -> LDS; overlap
  //      with the bf16 conversion staging below, drain via s_wait_asynccnt. ----
#if HAVE_ASYNC_LDS
  if (t < 64) {                                  // W2: 4 rows x 64 f32, 16B chunks
    const int n = t >> 4, c = t & 15;
    float* gp = const_cast<float*>(W2 + ((size_t)g * Oo + n) * Hh + c * 4);
    __builtin_amdgcn_global_load_async_to_lds_b128(
        (AS1 i32x4*)(void*)gp, (AS3 i32x4*)(void*)&w2s[n * W2S + c * 4], 0, 0);
  } else if (t < 80) {                           // b1: 64 f32
    const int c = t - 64;
    float* gp = const_cast<float*>(b1 + (size_t)g * Hh + c * 4);
    __builtin_amdgcn_global_load_async_to_lds_b128(
        (AS1 i32x4*)(void*)gp, (AS3 i32x4*)(void*)&b1s[c * 4], 0, 0);
  } else if (t == 80) {                          // b2: 4 f32 = one B128
    float* gp = const_cast<float*>(b2 + (size_t)g * Oo);
    __builtin_amdgcn_global_load_async_to_lds_b128(
        (AS1 i32x4*)(void*)gp, (AS3 i32x4*)(void*)&b2s[0], 0, 0);
  }
#else
  if (t < 64) {
    const int n = t >> 4, c = (t & 15) * 4;
    float4 q = *reinterpret_cast<const float4*>(W2 + ((size_t)g * Oo + n) * Hh + c);
    w2s[n * W2S + c + 0] = q.x; w2s[n * W2S + c + 1] = q.y;
    w2s[n * W2S + c + 2] = q.z; w2s[n * W2S + c + 3] = q.w;
  }
  if (t < Hh) b1s[t] = b1[(size_t)g * Hh + t];
  if (t < Oo) b2s[t] = b2[(size_t)g * Oo + t];
#endif
  // Zero the fc2 N-pad rows (4..15) -- disjoint from the DMA destinations.
  for (int idx = t; idx < 12 * Hh; idx += 256) {
    const int n = 4 + (idx >> 6);
    const int k = idx & 63;
    w2s[n * W2S + k] = 0.f;
  }

  __builtin_prefetch(items + ((size_t)(t >> 1) * Gg + g) * Vv, 0, 0); // global_prefetch_b8

  // ---- Stage X = [day | items[:,g,:] | 0pad], split fp32 -> bf16 hi + lo ----
  {
    const int row  = t >> 1;
    const int half = t & 1;
    const float* itg = items + ((size_t)row * Gg + g) * Vv;
    float vals[32];
    if (half == 0) {                       // cols 0..31 = [day(16) | items 0..15]
      ld16f(vals,      day + row * Cc);
      ld16f(vals + 16, itg);
    } else {                               // cols 32..63 = [items 16..31 | zeros]
      ld16f(vals,      itg + 16);
#pragma unroll
      for (int j = 0; j < 16; ++j) vals[16 + j] = 0.f;
    }
    const int cb = half * 32;
#pragma unroll
    for (int j = 0; j < 32; ++j) {
      float  f  = vals[j];
      __bf16 hi = (__bf16)f;
      __bf16 lo = (__bf16)(f - (float)hi);
      Xh[row * XK + cb + j] = hi;
      Xl[row * XK + cb + j] = lo;
    }
  }

  // ---- Stage fc1 weights as B-matrix (N-major [n][k]), hi + lo ----
  {
    const int n  = t & 63;
    const int kb = (t >> 6) * 16;          // 0,16,32,48
    float vals[16];
    if (kb == 0)       ld16f(vals, W1_day + ((size_t)g * Hh + n) * Cc);
    else if (kb == 16) ld16f(vals, W1_var + ((size_t)g * Hh + n) * Vv);
    else if (kb == 32) ld16f(vals, W1_var + ((size_t)g * Hh + n) * Vv + 16);
    else {
#pragma unroll
      for (int j = 0; j < 16; ++j) vals[j] = 0.f;    // K pad 48..63
    }
#pragma unroll
    for (int j = 0; j < 16; ++j) {
      float  f  = vals[j];
      __bf16 hi = (__bf16)f;
      __bf16 lo = (__bf16)(f - (float)hi);
      Wh[n * XK + kb + j] = hi;
      Wl[n * XK + kb + j] = lo;
    }
  }

#if HAVE_ASYNC_LDS
#if __has_builtin(__builtin_amdgcn_s_wait_asynccnt)
  __builtin_amdgcn_s_wait_asynccnt(0);
#else
  asm volatile("s_wait_asynccnt 0" ::: "memory");
#endif
#endif
  __syncthreads();

  // ---- fc1: compensated bf16 WMMA (Ahi*Whi + Alo*Whi + Ahi*Wlo), fp32 accum ----
  const int m = wave;
  Frag16 Ah[2], Al[2];
  {
    // A 16x32 bf16 layout: lanes 0-15 hold K{kb..kb+7, kb+16..kb+23}, kb=8*hsel
    const __bf16* xh = &Xh[(size_t)(16 * m + l15) * XK];
    const __bf16* xl = &Xl[(size_t)(16 * m + l15) * XK];
#pragma unroll
    for (int s = 0; s < 2; ++s) {
      const int kb = 32 * s + 8 * hsel;
      Ah[s].u[0] = *reinterpret_cast<const u32x4*>(xh + kb);
      Ah[s].u[1] = *reinterpret_cast<const u32x4*>(xh + kb + 16);
      Al[s].u[0] = *reinterpret_cast<const u32x4*>(xl + kb);
      Al[s].u[1] = *reinterpret_cast<const u32x4*>(xl + kb + 16);
    }
  }

#pragma unroll
  for (int nt = 0; nt < 4; ++nt) {
    const int col = 16 * nt + l15;
    // B 32x16 bf16 layout: column = l15, K{kb..kb+7, kb+16..kb+23}, kb = 32*s + 8*hsel
    Frag16 Bh[2], Bl[2];
#pragma unroll
    for (int s = 0; s < 2; ++s) {
      const __bf16* wh = &Wh[(size_t)col * XK + 32 * s + 8 * hsel];
      const __bf16* wl = &Wl[(size_t)col * XK + 32 * s + 8 * hsel];
      Bh[s].u[0] = *reinterpret_cast<const u32x4*>(wh);
      Bh[s].u[1] = *reinterpret_cast<const u32x4*>(wh + 16);
      Bl[s].u[0] = *reinterpret_cast<const u32x4*>(wl);
      Bl[s].u[1] = *reinterpret_cast<const u32x4*>(wl + 16);
    }
    const float bias = b1s[col];
    v8f acc = {bias, bias, bias, bias, bias, bias, bias, bias};
#pragma unroll
    for (int s = 0; s < 2; ++s) {
      acc = __builtin_amdgcn_wmma_f32_16x16x32_bf16(false, Ah[s].v, false, Bh[s].v,
                                                    (short)0, acc, false, false);
      acc = __builtin_amdgcn_wmma_f32_16x16x32_bf16(false, Al[s].v, false, Bh[s].v,
                                                    (short)0, acc, false, false);
      acc = __builtin_amdgcn_wmma_f32_16x16x32_bf16(false, Ah[s].v, false, Bl[s].v,
                                                    (short)0, acc, false, false);
    }
    // ReLU + stash hidden (C/D layout: VGPR r <-> row 16m + r + 8*hsel)
#pragma unroll
    for (int r = 0; r < 8; ++r) {
      float hv = acc[r];
      hv = hv > 0.f ? hv : 0.f;
      hbuf[(size_t)(16 * m + r + 8 * hsel) * HS + col] = hv;
    }
  }
  __syncthreads();

  // ---- fc2: exact fp32 via V_WMMA_F32_16X16X4_F32 (N padded 4->16, 16 k-steps) ----
  {
    const int col = l15;
    const float bias2 = (col < Oo) ? b2s[col] : 0.f;
    v8f acc2 = {bias2, bias2, bias2, bias2, bias2, bias2, bias2, bias2};
    const float* hrow = &hbuf[(size_t)(16 * m + l15) * HS];  // A row = M tile row
    const float* wrow = &w2s[col * W2S];                      // B col = l15
#pragma unroll
    for (int s = 0; s < 16; ++s) {
      const int k = 4 * s + 2 * hsel;       // f32 A 16x4 / B 4x16: half-wave K split
      v2f a = {hrow[k], hrow[k + 1]};
      v2f b = {wrow[k], wrow[k + 1]};
      acc2 = __builtin_amdgcn_wmma_f32_16x16x4_f32(false, a, false, b,
                                                   (short)0, acc2, false, false);
    }
    if (col < Oo) {
#pragma unroll
      for (int r = 0; r < 8; ++r) {
        const int row = 16 * m + r + 8 * hsel;
        out[(size_t)row * (Gg * Oo) + (size_t)g * Oo + col] = acc2[r];
      }
    }
  }
}

extern "C" void kernel_launch(void* const* d_in, const int* in_sizes, int n_in,
                              void* d_out, int out_size, void* d_ws, size_t ws_size,
                              hipStream_t stream) {
  const float* day    = (const float*)d_in[0];
  const float* items  = (const float*)d_in[1];
  const float* W1_day = (const float*)d_in[2];
  const float* W1_var = (const float*)d_in[3];
  const float* b1     = (const float*)d_in[4];
  const float* W2     = (const float*)d_in[5];
  const float* b2     = (const float*)d_in[6];
  float* out = (float*)d_out;
  (void)in_sizes; (void)n_in; (void)out_size; (void)d_ws; (void)ws_size;

  splitmlp_wmma<<<dim3(Gg), dim3(256), 0, stream>>>(
      day, items, W1_day, W1_var, b1, W2, b2, out);
}
